// Decoder_27487790694752
// MI455X (gfx1250) — matrix operand order repaired and checked by reading, not compiled, not run on hardware
//
#include <hip/hip_runtime.h>
#include <hip/hip_bf16.h>
#include <cstdint>
#include <cstddef>

// ---------------------------------------------------------------------------
// MI455X (gfx1250) implementation: bf16 WMMA GEMMs + elementwise LSTM cells.
// Dims (fixed by reference): B=256 T=512 Z=512 CH=1024 LH=1024 S=16 N=16.
// Workspace use: ~231 MB (weights bf16 43MB, pre_ctx 64MB, gates 64MB, misc).
// ---------------------------------------------------------------------------

typedef __bf16 bf16;
typedef __attribute__((ext_vector_type(16))) __bf16 v16bf;
typedef __attribute__((ext_vector_type(8)))  __bf16 v8bf;
typedef __attribute__((ext_vector_type(8)))  float  v8f;
typedef __attribute__((ext_vector_type(4)))  int    v4i;

union V16U { v16bf v; v8bf h[2]; };

__device__ __forceinline__ bf16 f2bf(float x) {
    union { float f; unsigned u; } a; a.f = x;
    unsigned r = a.u + 0x7FFFu + ((a.u >> 16) & 1u);   // round-to-nearest-even
    unsigned short hs = (unsigned short)(r >> 16);
    return __builtin_bit_cast(bf16, hs);
}
__device__ __forceinline__ float sigm(float x) { return 1.0f / (1.0f + __expf(-x)); }
__device__ __forceinline__ float tanh_f(float x) {
    float e = __expf(2.0f * x);
    return 1.0f - 2.0f / (e + 1.0f);   // saturates correctly at +/-1
}

// ---- gfx1250 async global->LDS fill (ASYNCcnt-tracked), guarded -----------
#if defined(__HIP_DEVICE_COMPILE__) && __has_builtin(__builtin_amdgcn_global_load_async_to_lds_b128)
#define ASYNC_LDS 1
__device__ __forceinline__ void async_cp16(const bf16* g, bf16* l) {
    // Param types per clang diagnostic: (int4 AS1*, int4 AS3*, imm, imm).
    // AS1 = numeric generic global address; AS3 = low 32 bits of generic LDS
    // address.
    __attribute__((address_space(1))) v4i* gp =
        (__attribute__((address_space(1))) v4i*)(uintptr_t)g;
    __attribute__((address_space(3))) v4i* lp =
        (__attribute__((address_space(3))) v4i*)(uint32_t)(uintptr_t)l;
    __builtin_amdgcn_global_load_async_to_lds_b128(gp, lp, 0, 0);
}
__device__ __forceinline__ void wait_async0() {
#if __has_builtin(__builtin_amdgcn_s_wait_asynccnt)
    __builtin_amdgcn_s_wait_asynccnt(0);
#else
    asm volatile("s_wait_asynccnt 0" ::: "memory");
#endif
}
#else
#define ASYNC_LDS 0
#endif

// ---------------------------------------------------------------------------
// Generic NT GEMM: C[M,N] = act( A[M,K] @ W[N,K]^T + bias + addsrc )
// A, W: bf16 row-major (K contiguous). C: f32.
// scatter_n >= 0: row m=(s*256+b) is written to output row b*256 + s*16 + n
//                 (the logits (N,S,B,T) -> (B,S*N,T) transpose, fused).
// Requires: M%128==0, N%128==0, K%64==0 (true for every call below).
// ---------------------------------------------------------------------------
#define BM 128
#define BN 128
#define BK 64
#define KST (BK + 8)   // padded LDS row stride (144 B, 16B-aligned)

__global__ __launch_bounds__(256)
void wmma_gemm(const bf16* __restrict__ A, int lda,
               const bf16* __restrict__ W, int ldw,
               float* __restrict__ C, int ldc,
               const float* __restrict__ bias,
               const float* __restrict__ addsrc, int ldadd,
               int M, int Nn, int K, int act, int scatter_n)
{
    __shared__ __align__(16) bf16 As[2][BM][KST];
    __shared__ __align__(16) bf16 Bs[2][BN][KST];

    const int tid    = threadIdx.x;
    const int lane   = tid & 31;
    const int wave   = tid >> 5;
    const int half   = lane >> 4;     // K-half selector per ISA layout
    const int lrow   = lane & 15;
    const int wmBase = (wave & 1) * 64;
    const int wnBase = (wave >> 1) * 32;
    const int blockM = blockIdx.y * BM;
    const int blockN = blockIdx.x * BN;

    // cooperative tile fill: each thread moves 32 bf16 (64B) per operand
    const int ldrow = tid >> 1;           // 0..127
    const int ldcol = (tid & 1) * 32;     // 0 or 32
    const bf16* Abase = A + (size_t)(blockM + ldrow) * lda + ldcol;
    const bf16* Wbase = W + (size_t)(blockN + ldrow) * ldw + ldcol;

    v8f acc[4][2];
#pragma unroll
    for (int mt = 0; mt < 4; ++mt)
#pragma unroll
        for (int nt = 0; nt < 2; ++nt)
#pragma unroll
            for (int v = 0; v < 8; ++v) acc[mt][nt][v] = 0.0f;

    const int nkt = K / BK;

#if ASYNC_LDS
    // ---- async double-buffered pipeline: no VGPR staging ----
    {
        bf16* as = &As[0][ldrow][ldcol];
        bf16* bs = &Bs[0][ldrow][ldcol];
#pragma unroll
        for (int i = 0; i < 4; ++i) {
            async_cp16(Abase + i * 8, as + i * 8);
            async_cp16(Wbase + i * 8, bs + i * 8);
        }
        wait_async0();
    }
    __syncthreads();
#else
    v8bf ar[4], br[4];
    {
        const v8bf* ap = (const v8bf*)Abase;
        const v8bf* wp = (const v8bf*)Wbase;
        ar[0]=ap[0]; ar[1]=ap[1]; ar[2]=ap[2]; ar[3]=ap[3];
        br[0]=wp[0]; br[1]=wp[1]; br[2]=wp[2]; br[3]=wp[3];
        v8bf* as = (v8bf*)&As[0][ldrow][ldcol];
        v8bf* bs = (v8bf*)&Bs[0][ldrow][ldcol];
        as[0]=ar[0]; as[1]=ar[1]; as[2]=ar[2]; as[3]=ar[3];
        bs[0]=br[0]; bs[1]=br[1]; bs[2]=br[2]; bs[3]=br[3];
    }
    __syncthreads();
#endif

    int buf = 0;
    for (int kt = 0; kt < nkt; ++kt) {
        const bool hasNext = (kt + 1 < nkt);

#if ASYNC_LDS
        if (hasNext) {   // fill buf^1 while computing from buf
            const bf16* an = Abase + (size_t)(kt + 1) * BK;
            const bf16* wn = Wbase + (size_t)(kt + 1) * BK;
            bf16* as = &As[buf ^ 1][ldrow][ldcol];
            bf16* bs = &Bs[buf ^ 1][ldrow][ldcol];
#pragma unroll
            for (int i = 0; i < 4; ++i) {
                async_cp16(an + i * 8, as + i * 8);
                async_cp16(wn + i * 8, bs + i * 8);
            }
        }
#else
        if (hasNext) {
            const v8bf* ap = (const v8bf*)(Abase + (size_t)(kt + 1) * BK);
            const v8bf* wp = (const v8bf*)(Wbase + (size_t)(kt + 1) * BK);
            ar[0]=ap[0]; ar[1]=ap[1]; ar[2]=ap[2]; ar[3]=ap[3];
            br[0]=wp[0]; br[1]=wp[1]; br[2]=wp[2]; br[3]=wp[3];
        }
#endif
        if (kt + 2 < nkt) {   // speculative L2 prefetch (global_prefetch_b8)
            __builtin_prefetch(Abase + (size_t)(kt + 2) * BK, 0, 1);
            __builtin_prefetch(Wbase + (size_t)(kt + 2) * BK, 0, 1);
        }

        // ---- compute from LDS[buf]: preload all fragments, then 8 WMMAs ----
#pragma unroll
        for (int ks = 0; ks < BK; ks += 32) {
            V16U afr[4], bfr[2];
#pragma unroll
            for (int nt = 0; nt < 2; ++nt) {
                // B 32x16: lane holds W-row (n = lane&15); lane<16 -> K{0..7,16..23}; lane>=16 -> K{8..15,24..31}
                const bf16* bp = &Bs[buf][wnBase + nt*16 + lrow][ks + half*8];
                bfr[nt].h[0] = *(const v8bf*)bp;
                bfr[nt].h[1] = *(const v8bf*)(bp + 16);
            }
#pragma unroll
            for (int mt = 0; mt < 4; ++mt) {
                // A 16x32: lane<16 -> K{0..7,16..23}; lane>=16 -> K{8..15,24..31}
                const bf16* ap = &As[buf][wmBase + mt*16 + lrow][ks + half*8];
                afr[mt].h[0] = *(const v8bf*)ap;
                afr[mt].h[1] = *(const v8bf*)(ap + 16);
            }
#pragma unroll
            for (int mt = 0; mt < 4; ++mt)
#pragma unroll
                for (int nt = 0; nt < 2; ++nt)
                    acc[mt][nt] = __builtin_amdgcn_wmma_f32_16x16x32_bf16(
                        false, afr[mt].v, false, bfr[nt].v,
                        (short)0, acc[mt][nt], false, false);
        }

#if ASYNC_LDS
        if (hasNext) wait_async0();   // my fills of buf^1 landed in LDS
        __syncthreads();              // everyone's fills visible; reads of buf done
#else
        if (hasNext) {
            v8bf* as = (v8bf*)&As[buf ^ 1][ldrow][ldcol];
            v8bf* bs = (v8bf*)&Bs[buf ^ 1][ldrow][ldcol];
            as[0]=ar[0]; as[1]=ar[1]; as[2]=ar[2]; as[3]=ar[3];
            bs[0]=br[0]; bs[1]=br[1]; bs[2]=br[2]; bs[3]=br[3];
        }
        __syncthreads();
#endif
        buf ^= 1;
    }

    // epilogue (C/D layout: lane = N col, VGPR v = M row (+8 for upper lanes))
#pragma unroll
    for (int mt = 0; mt < 4; ++mt) {
#pragma unroll
        for (int nt = 0; nt < 2; ++nt) {
            const int col = blockN + wnBase + nt*16 + lrow;
            const float bv = bias ? bias[col] : 0.0f;
#pragma unroll
            for (int v = 0; v < 8; ++v) {
                const int row = blockM + wmBase + mt*16 + half*8 + v;
                float xv = acc[mt][nt][v] + bv;
                if (addsrc) xv += addsrc[(size_t)row * ldadd + col];
                if (act) xv = tanh_f(xv);
                size_t drow = (size_t)row;
                if (scatter_n >= 0) {
                    const int s = row >> 8;        // M index = s*256 + b
                    const int b = row & 255;
                    drow = (size_t)(b * 256 + s * 16 + scatter_n);
                }
                C[drow * (size_t)ldc + col] = xv;
            }
        }
    }
}

// ---------------------------------------------------------------------------
// Elementwise helpers
// ---------------------------------------------------------------------------
__global__ void cvt_f32_bf16_k(const float* __restrict__ s, bf16* __restrict__ d, size_t n) {
    size_t i = (size_t)blockIdx.x * blockDim.x + threadIdx.x;
    if (i < n) d[i] = f2bf(s[i]);
}
__global__ void add_vec_k(const float* __restrict__ a, const float* __restrict__ b,
                          float* __restrict__ c, int n) {
    int i = blockIdx.x * blockDim.x + threadIdx.x;
    if (i < n) c[i] = a[i] + b[i];
}
// hc0 (M x 2H, already tanh'ed) -> h bf16 (first H), c f32 (second H)
__global__ void split_hc0_k(const float* __restrict__ src, bf16* __restrict__ h,
                            float* __restrict__ c, int M, int H) {
    int i = blockIdx.x * blockDim.x + threadIdx.x;
    if (i >= M * H) return;
    int m = i / H, j = i - m * H;
    const float* r = src + (size_t)m * 2 * H;
    h[i] = f2bf(r[j]);
    c[i] = r[H + j];
}
// dec_in (M x 2H f32) -> full bf16 copy + c f32 from second half
__global__ void split_decin_k(const float* __restrict__ src, bf16* __restrict__ full,
                              float* __restrict__ c, int M, int H) {
    size_t i = (size_t)blockIdx.x * blockDim.x + threadIdx.x;
    if (i >= (size_t)M * 2 * H) return;
    float v = src[i];
    full[i] = f2bf(v);
    int m = (int)(i / (2 * H));
    int j = (int)(i - (size_t)m * 2 * H);
    if (j >= H) c[(size_t)m * H + (j - H)] = v;
}
// tok_t for decoder step n: tok[(s*256+b)*512 + t] = (n==0) ? 0 : x[b, s*16+n-1, t]
__global__ void tok_build_k(const float* __restrict__ x, bf16* __restrict__ tok, int n) {
    int i = blockIdx.x * blockDim.x + threadIdx.x;       // over 4096*512
    if (i >= 4096 * 512) return;
    int t = i & 511;
    int m = i >> 9;
    int s = m >> 8;
    int b = m & 255;
    float v = 0.0f;
    if (n > 0) v = x[(size_t)b * (256 * 512) + (size_t)(s * 16 + (n - 1)) * 512 + t];
    tok[i] = f2bf(v);
}
// LSTM cell: gates (M x 4H, order i,f,g,o), c in/out f32, h out bf16
__global__ void lstm_cell_k(const float* __restrict__ gates, const float* __restrict__ cin,
                            float* __restrict__ cout, bf16* __restrict__ h, int M, int H) {
    size_t i = (size_t)blockIdx.x * blockDim.x + threadIdx.x;
    if (i >= (size_t)M * H) return;
    int m = (int)(i / H);
    int j = (int)(i - (size_t)m * H);
    const float* g = gates + (size_t)m * 4 * H;
    float ig = sigm(g[j]);
    float fg = sigm(g[H + j]);
    float gg = tanh_f(g[2 * H + j]);
    float og = sigm(g[3 * H + j]);
    float c = fg * cin[i] + ig * gg;
    cout[i] = c;
    h[i] = f2bf(og * tanh_f(c));
}

// ---------------------------------------------------------------------------
// Orchestration
// ---------------------------------------------------------------------------
extern "C" void kernel_launch(void* const* d_in, const int* in_sizes, int n_in,
                              void* d_out, int out_size, void* d_ws, size_t ws_size,
                              hipStream_t stream) {
    (void)in_sizes; (void)n_in; (void)out_size; (void)ws_size;
    const int Bb = 256, T = 512, Zc = 512, CH = 1024, LH = 1024, S = 16, Nc = 16;
    const int SB = S * Bb;                 // 4096

    const float* z        = (const float*)d_in[0];
    const float* x        = (const float*)d_in[1];
    const float* lin_in_w = (const float*)d_in[2];
    const float* lin_in_b = (const float*)d_in[3];
    // d_in[4] = cond_Wih: unused by reference
    const float* cond_Whh = (const float*)d_in[5];
    const float* cond_bih = (const float*)d_in[6];
    const float* cond_bhh = (const float*)d_in[7];
    const float* pre_w    = (const float*)d_in[8];
    const float* pre_b    = (const float*)d_in[9];
    const float* dec_Wih  = (const float*)d_in[10];
    const float* dec_Whh  = (const float*)d_in[11];
    const float* dec_bih  = (const float*)d_in[12];
    const float* dec_bhh  = (const float*)d_in[13];
    const float* out_w    = (const float*)d_in[14];
    const float* out_b    = (const float*)d_in[15];
    float* out = (float*)d_out;

    char* ws = (char*)d_ws;
    size_t off = 0;
    auto alloc = [&](size_t bytes) -> void* {
        void* p = ws + off;
        off += (bytes + 255) & ~(size_t)255;
        return p;
    };
    bf16*  w_lin     = (bf16*)alloc((size_t)2*CH * Zc * 2);
    bf16*  w_condWhh = (bf16*)alloc((size_t)4*CH * CH * 2);
    bf16*  w_pre     = (bf16*)alloc((size_t)2*LH * CH * 2);
    bf16*  w_decWih  = (bf16*)alloc((size_t)4*LH * (2*LH + T) * 2);
    bf16*  w_decWhh  = (bf16*)alloc((size_t)4*LH * LH * 2);
    bf16*  w_out     = (bf16*)alloc((size_t)T * LH * 2);
    bf16*  z_bf      = (bf16*)alloc((size_t)Bb * Zc * 2);
    float* cond_bias = (float*)alloc(4*CH * 4);
    float* dec_bias  = (float*)alloc(4*LH * 4);
    float* hc0       = (float*)alloc((size_t)Bb * 2*CH * 4);
    bf16*  hc_bf     = (bf16*)alloc((size_t)Bb * CH * 2);
    float* cc        = (float*)alloc((size_t)Bb * CH * 4);
    float* cgates    = (float*)alloc((size_t)Bb * 4*CH * 4);
    bf16*  cond_outs = (bf16*)alloc((size_t)SB * CH * 2);
    bf16*  decin_bf  = (bf16*)alloc((size_t)SB * 2*LH * 2);
    float* dc        = (float*)alloc((size_t)SB * LH * 4);
    float* pre_ctx   = (float*)alloc((size_t)SB * 4*LH * 4);
    float* dgates    = (float*)alloc((size_t)SB * 4*LH * 4);
    bf16*  tok_bf    = (bf16*)alloc((size_t)SB * T * 2);
    bf16*  h_bf      = (bf16*)alloc((size_t)SB * LH * 2);
    float* dec_in_f  = dgates;   // 32MB, overlaps dgates (used only pre-loop)

    auto cvt = [&](const float* s, bf16* d, size_t n) {
        cvt_f32_bf16_k<<<(unsigned)((n + 255) / 256), 256, 0, stream>>>(s, d, n);
    };
    auto gemm = [&](const bf16* A, int lda, const bf16* W, int ldw,
                    float* C, int ldc, const float* bias,
                    const float* addsrc, int ldadd,
                    int M, int Nn, int K, int act, int scatter) {
        dim3 g(Nn / BN, M / BM), b(256);
        wmma_gemm<<<g, b, 0, stream>>>(A, lda, W, ldw, C, ldc, bias,
                                       addsrc, ldadd, M, Nn, K, act, scatter);
    };

    // --- weight / input conversion ---
    cvt(lin_in_w, w_lin,     (size_t)2*CH * Zc);
    cvt(cond_Whh, w_condWhh, (size_t)4*CH * CH);
    cvt(pre_w,    w_pre,     (size_t)2*LH * CH);
    cvt(dec_Wih,  w_decWih,  (size_t)4*LH * (2*LH + T));
    cvt(dec_Whh,  w_decWhh,  (size_t)4*LH * LH);
    cvt(out_w,    w_out,     (size_t)T * LH);
    cvt(z,        z_bf,      (size_t)Bb * Zc);
    add_vec_k<<<(4*CH + 255) / 256, 256, 0, stream>>>(cond_bih, cond_bhh, cond_bias, 4*CH);
    add_vec_k<<<(4*LH + 255) / 256, 256, 0, stream>>>(dec_bih, dec_bhh, dec_bias, 4*LH);

    // --- hc0 = tanh(z @ lin_in_w^T + b); split into h(bf16), c(f32) ---
    gemm(z_bf, Zc, w_lin, Zc, hc0, 2*CH, lin_in_b, nullptr, 0, Bb, 2*CH, Zc, 1, -1);
    split_hc0_k<<<(Bb*CH + 255) / 256, 256, 0, stream>>>(hc0, hc_bf, cc, Bb, CH);

    // --- conditioning LSTM: S steps of h @ Whh^T + bias ---
    const bf16* hcur = hc_bf;
    for (int s = 0; s < S; ++s) {
        gemm(hcur, CH, w_condWhh, CH, cgates, 4*CH, cond_bias, nullptr, 0,
             Bb, 4*CH, CH, 0, -1);
        bf16* hnext = cond_outs + (size_t)s * Bb * CH;
        lstm_cell_k<<<(Bb*CH + 255) / 256, 256, 0, stream>>>(cgates, cc, cc, hnext, Bb, CH);
        hcur = hnext;
    }

    // --- dec_in = cond_outs @ pre_w^T + pre_b ; split into h/c ---
    gemm(cond_outs, CH, w_pre, CH, dec_in_f, 2*LH, pre_b, nullptr, 0,
         SB, 2*LH, CH, 0, -1);
    split_decin_k<<<(unsigned)(((size_t)SB*2*LH + 255) / 256), 256, 0, stream>>>(
        dec_in_f, decin_bf, dc, SB, LH);

    // --- pre_ctx = dec_in @ W_ctx^T + (dec_bih + dec_bhh)  [bias folded] ---
    gemm(decin_bf, 2*LH, w_decWih, 2*LH + T, pre_ctx, 4*LH, dec_bias, nullptr, 0,
         SB, 4*LH, 2*LH, 0, -1);

    // --- decoder LSTM: N steps; fused scatter-store of logits per step ---
    const bf16* hd = decin_bf;  int hd_ld = 2*LH;   // dh = dec_in[:, :LH]
    for (int n = 0; n < Nc; ++n) {
        tok_build_k<<<(SB*T + 255) / 256, 256, 0, stream>>>(x, tok_bf, n);
        // gates = pre_ctx + tok @ W_tok^T        (W_tok = dec_Wih[:, 2LH:])
        gemm(tok_bf, T, w_decWih + 2*LH, 2*LH + T, dgates, 4*LH,
             nullptr, pre_ctx, 4*LH, SB, 4*LH, T, 0, -1);
        // gates += h @ Whh^T
        gemm(hd, hd_ld, w_decWhh, LH, dgates, 4*LH,
             nullptr, dgates, 4*LH, SB, 4*LH, LH, 0, -1);
        lstm_cell_k<<<(unsigned)(((size_t)SB*LH + 255) / 256), 256, 0, stream>>>(
            dgates, dc, dc, h_bf, SB, LH);
        hd = h_bf; hd_ld = LH;
        // logits_n = h @ out_w^T + out_b, scattered to d_out[(b, s*16+n), :]
        gemm(h_bf, LH, w_out, LH, out, T, out_b, nullptr, 0, SB, T, LH, 0, n);
    }
}
